// AODNet_41815801594046
// MI455X (gfx1250) — matrix operand order repaired and checked by reading, not compiled, not run on hardware
//
#include <hip/hip_runtime.h>
#include <hip/hip_fp16.h>
#include <math.h>

typedef __attribute__((ext_vector_type(16))) _Float16 v16h;
typedef __attribute__((ext_vector_type(8)))  float    v8f;

#define BB 8
#define HH 1024
#define WW 1024
#define HW (HH*WW)

// CDNA5 async global->LDS copy availability (device pass only; host falls back)
#if defined(__gfx1250__) && __has_builtin(__builtin_amdgcn_global_load_async_to_lds_b32) && \
    __has_builtin(__builtin_amdgcn_s_wait_asynccnt)
#define HAVE_ASYNC_LDS 1
#else
#define HAVE_ASYNC_LDS 0
#endif

// ---------------- stats region layout (floats, relative to stats base) ----
#define S_BHIST 0                 // B*1024 brightness-bin counts
#define S_BSUM  (S_BHIST + BB*1024)       // B*1024*3 per-bin color sums
#define S_OVER  (S_BSUM  + BB*1024*3)     // B overall sums (r+g+b over image)
#define S_A     (S_OVER  + BB)            // B*3 atmospheric light
#define S_H256  (S_A     + BB*3)          // B*3*256 hist-eq histogram
#define S_LUT   (S_H256  + BB*3*256)      // B*3*256 LUT
#define S_MM    (S_LUT   + BB*3*256)      // B*3*2 min/max (float bits via uint atomics)
#define S_TOTAL (S_MM    + BB*3*2)

// ============================ init stats ==================================
__global__ void init_stats_kernel(float* stats) {
    int i = blockIdx.x * blockDim.x + threadIdx.x;
    if (i >= S_TOTAL) return;
    if (i >= S_MM) {
        int r = i - S_MM;
        // even slot = min (init +inf bits), odd slot = max (init 0)
        ((unsigned int*)stats)[i] = (r & 1) ? 0u : 0x7f800000u;
    } else {
        stats[i] = 0.0f;
    }
}

// ============================ clip input ==================================
__global__ void clip_kernel(const float* __restrict__ x, float* __restrict__ xc, size_t n) {
    size_t i = (size_t)blockIdx.x * blockDim.x + threadIdx.x;
    if (i < n) xc[i] = fminf(fmaxf(x[i], 0.0f), 1.0f);
}

// ===================== WMMA implicit-GEMM convolution =====================
// 256-thread block (8 waves) computes a 16x8 pixel tile. The block stages the
// (8+2p)x(16+2p)xCin input halo tile into LDS (async global->LDS on CDNA5),
// then each wave runs implicit-GEMM: M = out channels (padded to 16),
// N = 16 pixels of its row, K = Cin*kh*kw chunked by 32 into
// v_wmma_f32_16x16x32_f16 with f32 accumulation.
struct ConvSrcs {
    const float* p[4];
    int ch[4];
    int nsrc;
};

#define CT_TW 16
#define CT_TH 8
#define CT_MAXC 12
#define CT_MAXP 3

__global__ __launch_bounds__(256)
void conv_wmma_kernel(ConvSrcs srcs, const float* __restrict__ w,
                      const float* __restrict__ bias, float* __restrict__ out,
                      int cin, int cout, int kh, int kw, int pad, int act)
{
    __shared__ float wlds[1024];   // staged OIHW weights (max 882 for w4)
    __shared__ int   kmap[352];    // per-K (ci<<16 | dy<<8 | dx)
    __shared__ float tlds[CT_MAXC * (CT_TH + 2*CT_MAXP) * (CT_TW + 2*CT_MAXP)];

    const int tid  = threadIdx.x;
    const int lane = tid & 31;
    const int wrow = tid >> 5;             // wave id = output row within tile
    const int x0   = blockIdx.x << 4;
    const int y0   = blockIdx.y << 3;
    const int b    = blockIdx.z;

    const int kpix  = kh * kw;
    const int ktrue = cin * kpix;
    const int nw    = cout * ktrue;
    const int th    = CT_TH + 2 * pad;
    const int tw    = CT_TW + 2 * pad;

    for (int i = tid; i < nw; i += 256) wlds[i] = w[i];
    for (int k = tid; k < ktrue; k += 256) {
        int ci = k / kpix;
        int r  = k - ci * kpix;
        int dy = r / kw;
        int dx = r - dy * kw;
        kmap[k] = (ci << 16) | (dy << 8) | dx;
    }

    // ---- stage input halo tile into LDS ----
    const int total = cin * th * tw;
    for (int idx = tid; idx < total; idx += 256) {
        int ci  = idx / (th * tw);
        int rr  = idx - ci * (th * tw);
        int row = rr / tw;
        int col = rr - row * tw;
        int iy  = y0 - pad + row;
        int ix  = x0 - pad + col;
        float* dst = &tlds[idx];
        if (iy >= 0 && iy < HH && ix >= 0 && ix < WW) {
            int s = 0, cl = ci;
            while (s < srcs.nsrc - 1 && cl >= srcs.ch[s]) { cl -= srcs.ch[s]; ++s; }
            const float* g = &srcs.p[s][((size_t)b * srcs.ch[s] + cl) * (size_t)HW
                                        + (size_t)iy * WW + ix];
#if HAVE_ASYNC_LDS
            __builtin_amdgcn_global_load_async_to_lds_b32(
                (__attribute__((address_space(1))) int*)g,
                (__attribute__((address_space(3))) int*)dst, 0, 0);
#else
            *dst = *g;
#endif
        } else {
            *dst = 0.0f;
        }
    }
#if HAVE_ASYNC_LDS
    __builtin_amdgcn_s_wait_asynccnt(0);
#endif
    __syncthreads();

    const int m  = lane & 15;        // A row (out channel)
    const int n  = lane & 15;        // B col (pixel in tile row)
    const int hi = lane >> 4;        // upper/lower K half selector
    const int px = x0 + n;
    const int py = y0 + wrow;

    v8f acc = {};
    for (int kb = 0; kb < ktrue; kb += 32) {
        // ---- A fragment: 16x32 f16, row m, K layout per ISA 7.12.2 ----
        v16h afrag;
        #pragma unroll
        for (int i = 0; i < 16; ++i) {
            // lanes<16: idx0..7 -> K 0..7, idx8..15 -> K 16..23 ; lanes>=16: +8
            int kl = (i & 7) + ((i >> 3) << 4) + (hi << 3);
            int k  = kb + kl;
            float v = 0.0f;
            if (m < cout && k < ktrue) v = wlds[m * ktrue + k];
            afrag[i] = (_Float16)v;
        }
        // ---- B fragment: 32x16 f16 from LDS tile; same K layout as A (lanes<16 K 0..7,16..23; lanes>=16 +8) ----
        v16h bfrag;
        #pragma unroll
        for (int i = 0; i < 16; ++i) {
            int k = kb + (i & 7) + ((i >> 3) << 4) + (hi << 3);
            float v = 0.0f;
            if (k < ktrue) {
                int mp = kmap[k];
                int ci = mp >> 16;
                int dy = (mp >> 8) & 0xff;
                int dx = mp & 0xff;
                v = tlds[(ci * th + (wrow + dy)) * tw + (n + dx)];
            }
            bfrag[i] = (_Float16)v;
        }
        acc = __builtin_amdgcn_wmma_f32_16x16x32_f16(
                  false, afrag, false, bfrag, (short)0, acc, false, false);
    }

    // D layout: VGPR r holds (M = r + 8*hi, N = n)
    #pragma unroll
    for (int r = 0; r < 8; ++r) {
        int mrow = r + (hi << 3);
        if (mrow < cout) {
            float v = acc[r] + bias[mrow];
            if (act >= 1) v = fmaxf(v, 0.0f);
            if (act == 2) v = fminf(fmaxf(v, 0.05f), 2.0f);
            out[((size_t)b * cout + mrow) * (size_t)HW + (size_t)py * WW + px] = v;
        }
    }
}

// ================= atmospheric light: brightness histogram ================
__global__ void atmo_stats_kernel(const float* __restrict__ xc, float* stats) {
    __shared__ float cnt[1024];
    __shared__ float sums[3 * 1024];
    __shared__ float osum[256];

    const int tid = threadIdx.x;
    const int b = blockIdx.y;
    for (int i = tid; i < 1024; i += 256) cnt[i] = 0.0f;
    for (int i = tid; i < 3 * 1024; i += 256) sums[i] = 0.0f;
    __syncthreads();

    const float* img = xc + (size_t)b * 3 * HW;
    float local = 0.0f;
    for (int i = blockIdx.x * blockDim.x + tid; i < HW; i += gridDim.x * blockDim.x) {
        float r = img[i];
        float g = img[(size_t)HW + i];
        float bl = img[2 * (size_t)HW + i];
        float bright = (r + g + bl) * (1.0f / 3.0f);
        int bin = min(1023, max(0, (int)(bright * 1024.0f)));
        atomicAdd(&cnt[bin], 1.0f);
        atomicAdd(&sums[bin * 3 + 0], r);
        atomicAdd(&sums[bin * 3 + 1], g);
        atomicAdd(&sums[bin * 3 + 2], bl);
        local += r + g + bl;
    }
    osum[tid] = local;
    __syncthreads();
    for (int s = 128; s > 0; s >>= 1) {
        if (tid < s) osum[tid] += osum[tid + s];
        __syncthreads();
    }
    if (tid == 0) atomicAdd(&stats[S_OVER + b], osum[0]);
    for (int i = tid; i < 1024; i += 256)
        atomicAdd(&stats[S_BHIST + b * 1024 + i], cnt[i]);
    for (int i = tid; i < 3 * 1024; i += 256)
        atomicAdd(&stats[S_BSUM + b * 3072 + i], sums[i]);
}

__global__ void atmo_finalize_kernel(float* stats) {
    if (threadIdx.x != 0) return;
    const int b = blockIdx.x;
    const float nb = 1048.0f;   // int(1024*1024*0.001)
    float accC = 0.0f, accS0 = 0.0f, accS1 = 0.0f, accS2 = 0.0f;
    for (int bin = 1023; bin >= 0; --bin) {
        float c  = stats[S_BHIST + b * 1024 + bin];
        float s0 = stats[S_BSUM + b * 3072 + bin * 3 + 0];
        float s1 = stats[S_BSUM + b * 3072 + bin * 3 + 1];
        float s2 = stats[S_BSUM + b * 3072 + bin * 3 + 2];
        if (accC + c >= nb) {
            float fr = (c > 0.0f) ? (nb - accC) / c : 0.0f;
            accS0 += fr * s0; accS1 += fr * s1; accS2 += fr * s2;
            break;
        }
        accC += c; accS0 += s0; accS1 += s1; accS2 += s2;
    }
    float overall = stats[S_OVER + b] / (3.0f * (float)HW);
    float scale = fminf(fmaxf(1.2f - overall, 0.8f), 1.5f);
    stats[S_A + b * 3 + 0] = fminf(fmaxf(accS0 / nb * scale, 0.5f), 0.95f);
    stats[S_A + b * 3 + 1] = fminf(fmaxf(accS1 / nb * scale, 0.5f), 0.95f);
    stats[S_A + b * 3 + 2] = fminf(fmaxf(accS2 / nb * scale, 0.5f), 0.95f);
}

// ======================= scene radiance recovery ==========================
__device__ __forceinline__ float nan_fix(float v) {
    if (isnan(v)) return 0.5f;
    if (isinf(v)) return v > 0.0f ? 1.0f : 0.0f;
    return v;
}

__global__ void recover_kernel(const float* __restrict__ xc, const float* __restrict__ k,
                               const float* stats, float* __restrict__ J) {
    size_t i = (size_t)blockIdx.x * blockDim.x + threadIdx.x;
    if (i >= (size_t)BB * HW) return;
    int b = (int)(i / HW);
    size_t pix = i - (size_t)b * HW;
    size_t base = (size_t)b * 3 * HW + pix;
    float r = xc[base], g = xc[base + HW], bl = xc[base + 2 * (size_t)HW];
    float brightness = (r + g + bl) * (1.0f / 3.0f);
    float strength = fminf(fmaxf(1.0f - brightness, 0.3f), 0.8f);
    #pragma unroll
    for (int c = 0; c < 3; ++c) {
        float xv = xc[base + (size_t)c * HW];
        float kv = k[base + (size_t)c * HW];
        float A  = stats[S_A + b * 3 + c];
        float t  = fminf(fmaxf(1.0f - strength * kv, 0.1f), 1.0f);
        float j  = (xv - A) / (t + 1e-5f) + A;
        J[base + (size_t)c * HW] = nan_fix(j);
    }
}

// ================== blend: res = 0.5*(J+0.2*ref) + 0.5*xc =================
__global__ void blend_kernel(const float* __restrict__ J, const float* __restrict__ ref,
                             float* __restrict__ res, size_t n) {
    size_t i = (size_t)blockIdx.x * blockDim.x + threadIdx.x;
    if (i < n) res[i] = 0.5f * (J[i] + 0.2f * ref[i]) + 0.5f * res[i]; // res holds xc
}

// ========================= hist-eq: histogram =============================
__global__ void histeq_hist_kernel(const float* __restrict__ res, float* stats) {
    __shared__ float h[256];
    const int tid = threadIdx.x;
    const int row = blockIdx.y;          // b*3 + c
    h[tid] = 0.0f;
    __syncthreads();
    size_t base = (size_t)row * HW;
    for (int i = blockIdx.x * blockDim.x + tid; i < HW; i += gridDim.x * blockDim.x) {
        float v = res[base + i];
        if (v >= 0.0f && v <= 1.0f) {
            int bin = min(255, max(0, (int)floorf(v * 256.0f)));
            atomicAdd(&h[bin], 1.0f);
        }
    }
    __syncthreads();
    atomicAdd(&stats[S_H256 + row * 256 + tid], h[tid]);
}

__global__ void histeq_lut_kernel(float* stats) {
    __shared__ float s[256];
    const int tid = threadIdx.x;
    const int row = blockIdx.x;
    s[tid] = stats[S_H256 + row * 256 + tid];
    __syncthreads();
    for (int off = 1; off < 256; off <<= 1) {
        float add = (tid >= off) ? s[tid - off] : 0.0f;
        __syncthreads();
        s[tid] += add;
        __syncthreads();
    }
    float total = s[255];
    stats[S_LUT + row * 256 + tid] = fminf(fmaxf(s[tid] / total, 0.0f), 1.0f);
}

__global__ void histeq_apply_kernel(float* __restrict__ res, float* stats) {
    __shared__ float smn[256], smx[256];
    const int tid = threadIdx.x;
    const int row = blockIdx.y;
    size_t base = (size_t)row * HW;
    const float* lut = stats + S_LUT + row * 256;
    float mn = 1e30f, mx = -1e30f;
    for (int i = blockIdx.x * blockDim.x + tid; i < HW; i += gridDim.x * blockDim.x) {
        float v = res[base + i];
        int li = min(255, max(0, (int)(v * 255.0f)));
        float r = lut[li];
        res[base + i] = r;
        mn = fminf(mn, r);
        mx = fmaxf(mx, r);
    }
    smn[tid] = mn; smx[tid] = mx;
    __syncthreads();
    for (int s = 128; s > 0; s >>= 1) {
        if (tid < s) { smn[tid] = fminf(smn[tid], smn[tid + s]);
                       smx[tid] = fmaxf(smx[tid], smx[tid + s]); }
        __syncthreads();
    }
    if (tid == 0) {     // results are in [0,1] -> float bit pattern is monotonic
        atomicMin((unsigned int*)stats + S_MM + row * 2,     __float_as_uint(smn[0]));
        atomicMax((unsigned int*)stats + S_MM + row * 2 + 1, __float_as_uint(smx[0]));
    }
}

// ============== stretch + saturation + sigmoid curve -> d_out =============
__global__ void final_kernel(const float* __restrict__ res, const float* stats,
                             float* __restrict__ out) {
    size_t i = (size_t)blockIdx.x * blockDim.x + threadIdx.x;
    if (i >= (size_t)BB * HW) return;
    int b = (int)(i / HW);
    size_t pix = i - (size_t)b * HW;
    size_t base = (size_t)b * 3 * HW + pix;
    float v[3];
    #pragma unroll
    for (int c = 0; c < 3; ++c) {
        float r = res[base + (size_t)c * HW];
        int row = b * 3 + c;
        float mn = __uint_as_float(((const unsigned int*)stats)[S_MM + row * 2]);
        float mx = __uint_as_float(((const unsigned int*)stats)[S_MM + row * 2 + 1]);
        float d = mx - mn;
        if (d > 0.05f) r = (r - mn) / fmaxf(d, 0.05f) * 0.95f;
        v[c] = r;
    }
    float mean = (v[0] + v[1] + v[2]) * (1.0f / 3.0f);
    #pragma unroll
    for (int c = 0; c < 3; ++c) {
        float s = mean + 1.3f * (v[c] - mean);
        float z = (s - 0.5f) * 5.0f;
        float sg = 0.95f / (1.0f + expf(-z)) + 0.025f;
        out[base + (size_t)c * HW] = fminf(fmaxf(sg, 0.0f), 1.0f);
    }
}

// ============================== launch ====================================
extern "C" void kernel_launch(void* const* d_in, const int* in_sizes, int n_in,
                              void* d_out, int out_size, void* d_ws, size_t ws_size,
                              hipStream_t stream) {
    const float* x   = (const float*)d_in[0];
    const float* w1  = (const float*)d_in[1];  const float* b1 = (const float*)d_in[2];
    const float* w2  = (const float*)d_in[3];  const float* b2 = (const float*)d_in[4];
    const float* w3  = (const float*)d_in[5];  const float* b3 = (const float*)d_in[6];
    const float* w4  = (const float*)d_in[7];  const float* b4 = (const float*)d_in[8];
    const float* w5  = (const float*)d_in[9];  const float* b5 = (const float*)d_in[10];
    const float* rw1 = (const float*)d_in[11]; const float* rb1 = (const float*)d_in[12];
    const float* rw2 = (const float*)d_in[13]; const float* rb2 = (const float*)d_in[14];

    const size_t SZ3 = (size_t)BB * 3 * HW;
    float* ws    = (float*)d_ws;
    float* xc    = ws;               // later: blended result (in place)
    float* x1    = ws + SZ3;
    float* x2    = ws + 2 * SZ3;
    float* kbuf  = ws + 3 * SZ3;
    float* x3    = ws + 4 * SZ3;     // later: J
    float* x4    = ws + 5 * SZ3;     // later: refined
    float* r1    = x1;               // 8ch, aliases dead x1/x2/kbuf (needs 8/9 of it)
    float* stats = ws + 6 * SZ3;

    auto mk1 = [](const float* p, int c) {
        ConvSrcs s{}; s.p[0] = p; s.ch[0] = c; s.nsrc = 1; return s;
    };
    auto mk2 = [](const float* p0, int c0, const float* p1, int c1) {
        ConvSrcs s{}; s.p[0] = p0; s.ch[0] = c0; s.p[1] = p1; s.ch[1] = c1; s.nsrc = 2; return s;
    };
    auto mk4 = [](const float* p0, const float* p1, const float* p2, const float* p3) {
        ConvSrcs s{}; s.p[0]=p0; s.p[1]=p1; s.p[2]=p2; s.p[3]=p3;
        s.ch[0]=s.ch[1]=s.ch[2]=s.ch[3]=3; s.nsrc = 4; return s;
    };

    dim3 cgrid(WW / CT_TW, HH / CT_TH, BB), cblk(256);
    int eb = 256;
    int gb3  = (int)((SZ3 + eb - 1) / eb);
    int gb1  = (int)(((size_t)BB * HW + eb - 1) / eb);

    init_stats_kernel<<<(S_TOTAL + eb - 1) / eb, eb, 0, stream>>>(stats);
    clip_kernel<<<gb3, eb, 0, stream>>>(x, xc, SZ3);

    // K-estimation trunk (implicit-GEMM on v_wmma_f32_16x16x32_f16)
    conv_wmma_kernel<<<cgrid, cblk, 0, stream>>>(mk1(xc, 3), w1, b1, x1, 3, 3, 1, 1, 0, 1);
    conv_wmma_kernel<<<cgrid, cblk, 0, stream>>>(mk1(x1, 3), w2, b2, x2, 3, 3, 3, 3, 1, 1);
    conv_wmma_kernel<<<cgrid, cblk, 0, stream>>>(mk2(x1, 3, x2, 3), w3, b3, x3, 6, 3, 5, 5, 2, 1);
    conv_wmma_kernel<<<cgrid, cblk, 0, stream>>>(mk2(x2, 3, x3, 3), w4, b4, x4, 6, 3, 7, 7, 3, 1);
    conv_wmma_kernel<<<cgrid, cblk, 0, stream>>>(mk4(x1, x2, x3, x4), w5, b5, kbuf, 12, 3, 3, 3, 1, 2);

    // atmospheric light
    atmo_stats_kernel<<<dim3(256, BB), 256, 0, stream>>>(xc, stats);
    atmo_finalize_kernel<<<BB, 32, 0, stream>>>(stats);

    // recovery J (overwrites x3)
    float* J = x3;
    recover_kernel<<<gb1, eb, 0, stream>>>(xc, kbuf, stats, J);

    // refinement residual (r1 aliases dead x1/x2/kbuf region)
    conv_wmma_kernel<<<cgrid, cblk, 0, stream>>>(mk1(J, 3), rw1, rb1, r1, 3, 8, 3, 3, 1, 1);
    float* refined = x4;
    conv_wmma_kernel<<<cgrid, cblk, 0, stream>>>(mk1(r1, 8), rw2, rb2, refined, 8, 3, 3, 3, 1, 0);

    // blend into xc (in place) -> res
    blend_kernel<<<gb3, eb, 0, stream>>>(J, refined, xc, SZ3);

    // histogram equalization + min/max
    histeq_hist_kernel<<<dim3(256, BB * 3), 256, 0, stream>>>(xc, stats);
    histeq_lut_kernel<<<BB * 3, 256, 0, stream>>>(stats);
    histeq_apply_kernel<<<dim3(256, BB * 3), 256, 0, stream>>>(xc, stats);

    // stretch + saturation + sigmoid
    final_kernel<<<gb1, eb, 0, stream>>>(xc, stats, (float*)d_out);
}
